// PatchSpatialAttention_2903397892245
// MI455X (gfx1250) — matrix operand order repaired and checked by reading, not compiled, not run on hardware
//
#include <hip/hip_runtime.h>
#include <hip/hip_bf16.h>
#include <math.h>

typedef __bf16 bf16;
typedef __attribute__((ext_vector_type(16))) __bf16 v16bf;
typedef __attribute__((ext_vector_type(8)))  float  v8f;

// Problem dims (fixed by reference)
constexpr int B_  = 8;
constexpr int P_  = 64;
constexpr int T_  = 512;
constexpr int D_  = 256;
constexpr int H_  = 8;
constexpr int DK_ = 32;
constexpr int FF_ = 512;
constexpr int NROWS = B_ * T_ * P_;   // 262144 rows of the (B,T,P,*) matrices

// row = (b*T + t)*P + p ;  T*P = 32768 = 1<<15, P = 64 = 1<<6
__device__ __forceinline__ void decode_row(int row, int& b, int& t, int& p) {
    b = row >> 15;
    t = (row >> 6) & (T_ - 1);
    p = row & (P_ - 1);
}

// ---------------------------------------------------------------------------
// WMMA fragment loaders (wave32, 16x16x32 bf16)
// A (16xK32): lane l -> m = l&15, half = l>>4; VGPR i<4: k=half*8+2i(+1); i>=4: k=16+half*8+2(i-4)
// B (K32x16) from LDS stored transposed [n][k]: lane l -> n = l&15, half = l>>4; VGPR i<4: k=half*8+2i(+1); i>=4: k=16+half*8+2(i-4)
// ---------------------------------------------------------------------------
__device__ __forceinline__ v16bf load_a_frag(const bf16* lds, int mBase, int stride,
                                             int kOff, int lane) {
    const int m    = mBase + (lane & 15);
    const int half = lane >> 4;
    v16bf a;
#pragma unroll
    for (int i = 0; i < 8; ++i) {
        int k = kOff + ((i < 4) ? (half * 8 + 2 * i) : (16 + half * 8 + 2 * (i - 4)));
        a[2 * i]     = lds[m * stride + k];
        a[2 * i + 1] = lds[m * stride + k + 1];
    }
    return a;
}

__device__ __forceinline__ v16bf load_bT_frag(const bf16* ldsT, int nBase, int stride,
                                              int kOff, int lane) {
    const int n    = nBase + (lane & 15);
    const int half = lane >> 4;
    v16bf bv;
#pragma unroll
    for (int i = 0; i < 8; ++i) {
        int k = kOff + ((i < 4) ? (half * 8 + 2 * i) : (16 + half * 8 + 2 * (i - 4)));
        bv[2 * i]     = ldsT[n * stride + k];
        bv[2 * i + 1] = ldsT[n * stride + k + 1];
    }
    return bv;
}

// ---------------------------------------------------------------------------
// Weight prep: fp32 (K x N, row-major) -> bf16 transposed (N x K, row-major)
// so GEMM B tiles load exactly like A tiles (vectorized, no LDS scatter).
// ---------------------------------------------------------------------------
__global__ void cvt_transpose_kernel(const float* __restrict__ s, bf16* __restrict__ d,
                                     int K, int N) {
    int i = blockIdx.x * 256 + threadIdx.x;
    if (i < K * N) {
        int k = i / N, n = i - k * N;
        d[(size_t)n * K + k] = (bf16)s[i];
    }
}

// ---------------------------------------------------------------------------
// Relative-position bias: rel_h (B,H,P,P) f32
// ---------------------------------------------------------------------------
__global__ __launch_bounds__(256) void relpos_kernel(const float* __restrict__ pos,
                                                     const float* __restrict__ wrel,
                                                     float* __restrict__ relh) {
    int i = blockIdx.x * 256 + threadIdx.x;   // < B*P*P = 32768
    int b = i >> 12;
    int r = i & 4095;
    int ii = r >> 6, jj = r & 63;
    const float d2r  = 0.017453292519943295f;
    float lat1 = pos[(b * P_ + ii) * 2 + 0] * d2r;
    float lon1 = pos[(b * P_ + ii) * 2 + 1] * d2r;
    float lat2 = pos[(b * P_ + jj) * 2 + 0] * d2r;
    float lon2 = pos[(b * P_ + jj) * 2 + 1] * d2r;
    float dlat = lat1 - lat2, dlon = lon1 - lon2;
    float dx = dlon * cosf(0.5f * (lat1 + lat2));
    float dy = dlat;
    float sdl = sinf(0.5f * dlat), sdo = sinf(0.5f * dlon);
    float a   = sdl * sdl + cosf(lat1) * cosf(lat2) * sdo * sdo;
    float sq  = sqrtf(fmaxf(a, 1e-12f));
    sq        = fminf(fmaxf(sq, 0.0f), 1.0f - 1e-7f);
    float dist = 2.0f * 6371.0f * asinf(sq);
    float bear = atan2f(sinf(dlon) * cosf(lat2),
                        cosf(lat1) * sinf(lat2) - sinf(lat1) * cosf(lat2) * cosf(dlon));
    float f0 = dx, f1 = dy, f2 = dist * (1.0f / 500.0f), f3 = sinf(bear);
#pragma unroll
    for (int hh = 0; hh < H_; ++hh) {
        float v = f0 * wrel[0 * H_ + hh] + f1 * wrel[1 * H_ + hh] +
                  f2 * wrel[2 * H_ + hh] + f3 * wrel[3 * H_ + hh];
        relh[(((size_t)b * H_ + hh) << 12) + r] = v;
    }
}

// ---------------------------------------------------------------------------
// LayerNorm 1: x (B,P,T,D) f32 -> xn (B,T,P,D) bf16. One wave per row.
// ---------------------------------------------------------------------------
__global__ __launch_bounds__(256) void ln1_kernel(const float* __restrict__ x,
                                                  const float* __restrict__ g,
                                                  const float* __restrict__ be,
                                                  bf16* __restrict__ xn) {
    const int lane = threadIdx.x & 31, w = threadIdx.x >> 5;
    const int row  = blockIdx.x * 8 + w;
    int b, t, p;
    decode_row(row, b, t, p);
    const float* src = x + (((size_t)(b * P_ + p)) * T_ + t) * D_;
    const int d0 = lane * 8;
    float v[8];
    const float4* sp = reinterpret_cast<const float4*>(src + d0);
    float4 f0 = sp[0], f1 = sp[1];
    v[0] = f0.x; v[1] = f0.y; v[2] = f0.z; v[3] = f0.w;
    v[4] = f1.x; v[5] = f1.y; v[6] = f1.z; v[7] = f1.w;
    float sum = 0.f, sq = 0.f;
#pragma unroll
    for (int i = 0; i < 8; ++i) { sum += v[i]; sq += v[i] * v[i]; }
    for (int m = 1; m < 32; m <<= 1) {
        sum += __shfl_xor(sum, m, 32);
        sq  += __shfl_xor(sq,  m, 32);
    }
    float mu  = sum * (1.0f / D_);
    float var = sq * (1.0f / D_) - mu * mu;
    float inv = rsqrtf(var + 1e-5f);
    bf16 outv[8];
#pragma unroll
    for (int i = 0; i < 8; ++i)
        outv[i] = (bf16)((v[i] - mu) * inv * g[d0 + i] + be[d0 + i]);
    *reinterpret_cast<uint4*>(xn + (size_t)row * D_ + d0) =
        *reinterpret_cast<const uint4*>(outv);
}

// ---------------------------------------------------------------------------
// LayerNorm 2 (in place on xt2 f32, also emits bf16 copy). One wave per row.
// ---------------------------------------------------------------------------
__global__ __launch_bounds__(256) void ln2_kernel(float* __restrict__ xt2,
                                                  const float* __restrict__ g,
                                                  const float* __restrict__ be,
                                                  bf16* __restrict__ ybf) {
    const int lane = threadIdx.x & 31, w = threadIdx.x >> 5;
    const int row  = blockIdx.x * 8 + w;
    float* src   = xt2 + (size_t)row * D_;
    const int d0 = lane * 8;
    float v[8];
    const float4* sp = reinterpret_cast<const float4*>(src + d0);
    float4 f0 = sp[0], f1 = sp[1];
    v[0] = f0.x; v[1] = f0.y; v[2] = f0.z; v[3] = f0.w;
    v[4] = f1.x; v[5] = f1.y; v[6] = f1.z; v[7] = f1.w;
    float sum = 0.f, sq = 0.f;
#pragma unroll
    for (int i = 0; i < 8; ++i) { sum += v[i]; sq += v[i] * v[i]; }
    for (int m = 1; m < 32; m <<= 1) {
        sum += __shfl_xor(sum, m, 32);
        sq  += __shfl_xor(sq,  m, 32);
    }
    float mu  = sum * (1.0f / D_);
    float var = sq * (1.0f / D_) - mu * mu;
    float inv = rsqrtf(var + 1e-5f);
    float  of[8];
    bf16   ob[8];
#pragma unroll
    for (int i = 0; i < 8; ++i) {
        float y = (v[i] - mu) * inv * g[d0 + i] + be[d0 + i];
        of[i] = y;
        ob[i] = (bf16)y;
    }
    float4* dp = reinterpret_cast<float4*>(src + d0);
    dp[0] = make_float4(of[0], of[1], of[2], of[3]);
    dp[1] = make_float4(of[4], of[5], of[6], of[7]);
    *reinterpret_cast<uint4*>(ybf + (size_t)row * D_ + d0) =
        *reinterpret_cast<const uint4*>(ob);
}

// ---------------------------------------------------------------------------
// Tiled bf16 WMMA GEMM: C = A(MxK) * B(KxN) + bias.  B supplied TRANSPOSED
// (N x K row-major).  Double-buffered LDS, one barrier per K-step, prefetch
// of the K-step-after-next (global_prefetch_b8).
// Block: 256 threads (8 waves), tile 128x128, K-step 32.
// Waves: 2(M) x 4(N); each wave 64x32 => 4x2 accumulators of 16x16.
// MODE 0: store bf16(acc+bias)                          (QKV)
// MODE 1: store bf16(gelu(acc+bias))                    (FF1)
// MODE 2: store f32(acc+bias + x[transposed row])       (out-proj + residual)
// MODE 3: store f32(acc+bias + resid) at transposed idx (FF2 + residual -> d_out)
// ---------------------------------------------------------------------------
template <int MODE>
__global__ __launch_bounds__(256) void gemm_bf16_kernel(
    const bf16* __restrict__ A, const bf16* __restrict__ Bt,
    const float* __restrict__ bias, int M, int N, int K,
    const float* __restrict__ resid, float* __restrict__ outF,
    bf16* __restrict__ outBf) {
    (void)M;
    __shared__ bf16 ldsA[2][128 * 32];
    __shared__ bf16 ldsB[2][128 * 32];   // [n][k]

    const int tid   = threadIdx.x;
    const int lane  = tid & 31;
    const int wid   = tid >> 5;
    const int waveM = wid & 1;    // 0..1 (64 rows each)
    const int waveN = wid >> 1;   // 0..3 (32 cols each)
    const int rowBase = blockIdx.y * 128;
    const int colBase = blockIdx.x * 128;

    v8f acc[4][2] = {};

    // identical staging pattern for A (M x K) and Bt (N x K)
    const int sm  = tid >> 1;          // 0..127 : row of A / row of Bt (== col of B)
    const int skk = (tid & 1) * 16;    // 0 / 16
    const bf16* aRow = A  + (size_t)(rowBase + sm) * K + skk;
    const bf16* bRow = Bt + (size_t)(colBase + sm) * K + skk;

    const int steps = K / 32;
    uint4 a0, a1, b0, b1;

    // preload K-step 0
    {
        const uint4* ap = reinterpret_cast<const uint4*>(aRow);
        const uint4* bp = reinterpret_cast<const uint4*>(bRow);
        a0 = ap[0]; a1 = ap[1];
        b0 = bp[0]; b1 = bp[1];
        *reinterpret_cast<uint4*>(&ldsA[0][sm * 32 + skk])     = a0;
        *reinterpret_cast<uint4*>(&ldsA[0][sm * 32 + skk + 8]) = a1;
        *reinterpret_cast<uint4*>(&ldsB[0][sm * 32 + skk])     = b0;
        *reinterpret_cast<uint4*>(&ldsB[0][sm * 32 + skk + 8]) = b1;
    }
    __syncthreads();

    for (int s = 0; s < steps; ++s) {
        const int cur = s & 1;
        if (s + 1 < steps) {
            const uint4* ap = reinterpret_cast<const uint4*>(aRow + (s + 1) * 32);
            const uint4* bp = reinterpret_cast<const uint4*>(bRow + (s + 1) * 32);
            a0 = ap[0]; a1 = ap[1];
            b0 = bp[0]; b1 = bp[1];
        }
        if (s + 2 < steps) {
            __builtin_prefetch(aRow + (s + 2) * 32, 0, 1);
            __builtin_prefetch(bRow + (s + 2) * 32, 0, 1);
        }

        v16bf af[4], bf[2];
#pragma unroll
        for (int mt = 0; mt < 4; ++mt)
            af[mt] = load_a_frag(ldsA[cur], waveM * 64 + mt * 16, 32, 0, lane);
#pragma unroll
        for (int nt = 0; nt < 2; ++nt)
            bf[nt] = load_bT_frag(ldsB[cur], waveN * 32 + nt * 16, 32, 0, lane);
#pragma unroll
        for (int mt = 0; mt < 4; ++mt)
#pragma unroll
            for (int nt = 0; nt < 2; ++nt)
                acc[mt][nt] = __builtin_amdgcn_wmma_f32_16x16x32_bf16(
                    false, af[mt], false, bf[nt], (short)0, acc[mt][nt], false, false);

        if (s + 1 < steps) {
            const int nxt = 1 - cur;
            *reinterpret_cast<uint4*>(&ldsA[nxt][sm * 32 + skk])     = a0;
            *reinterpret_cast<uint4*>(&ldsA[nxt][sm * 32 + skk + 8]) = a1;
            *reinterpret_cast<uint4*>(&ldsB[nxt][sm * 32 + skk])     = b0;
            *reinterpret_cast<uint4*>(&ldsB[nxt][sm * 32 + skk + 8]) = b1;
            __syncthreads();
        }
    }

    // Epilogue (C layout: VGPR r -> row r + 8*(lane>=16), col = lane&15)
    const int half = lane >> 4, cl = lane & 15;
#pragma unroll
    for (int mt = 0; mt < 4; ++mt)
#pragma unroll
        for (int nt = 0; nt < 2; ++nt)
#pragma unroll
            for (int r = 0; r < 8; ++r) {
                int row = rowBase + waveM * 64 + mt * 16 + r + 8 * half;
                int col = colBase + waveN * 32 + nt * 16 + cl;
                float v = acc[mt][nt][r] + bias[col];
                if constexpr (MODE == 0) {
                    outBf[(size_t)row * N + col] = (bf16)v;
                } else if constexpr (MODE == 1) {
                    float ge = 0.5f * v * (1.0f + erff(v * 0.70710678118654752f));
                    outBf[(size_t)row * N + col] = (bf16)ge;
                } else if constexpr (MODE == 2) {
                    int b, t, p;
                    decode_row(row, b, t, p);
                    float xr = resid[(((size_t)(b * P_ + p)) * T_ + t) * D_ + col];
                    outF[(size_t)row * N + col] = v + xr;
                } else {
                    float yr = resid[(size_t)row * N + col];
                    int b, t, p;
                    decode_row(row, b, t, p);
                    outF[(((size_t)(b * P_ + p)) * T_ + t) * D_ + col] = v + yr;
                }
            }
}

// ---------------------------------------------------------------------------
// Attention: one block (128 thr, 4 waves) per (b,t,h).
// S = q k^T * 1/sqrt(DK) + rel ; softmax over keys ; O = P v.
// ---------------------------------------------------------------------------
__global__ __launch_bounds__(128) void attn_kernel(const bf16* __restrict__ qkv,
                                                   const float* __restrict__ relh,
                                                   bf16* __restrict__ o) {
    __shared__ bf16 ldsQ[64 * 32];
    __shared__ bf16 ldsK[64 * 32];    // [p][d] == transposed-B storage for q*k^T
    __shared__ bf16 ldsVt[32 * 64];   // [dk][p] == transposed-B storage for P*v
    __shared__ bf16 ldsP[64 * 64];

    const int tid = threadIdx.x, lane = tid & 31, w = tid >> 5;
    const int bi = blockIdx.x;
    const int h  = bi & (H_ - 1);
    const int t  = (bi >> 3) & (T_ - 1);
    const int b  = bi >> 12;
    const size_t rowBase = ((size_t)b * T_ + t) * P_;
    const int    off     = h * DK_;

    // Load q/k/v 64x32 tiles (paired bf16 as u32).
#pragma unroll
    for (int it = 0; it < 8; ++it) {
        int idx = it * 128 + tid;       // 0..1023
        int p   = idx >> 4;
        int d2  = (idx & 15) * 2;
        const bf16* base = qkv + (rowBase + p) * (3 * D_) + off + d2;
        unsigned qv = *reinterpret_cast<const unsigned*>(base);
        unsigned kv = *reinterpret_cast<const unsigned*>(base + D_);
        unsigned vv = *reinterpret_cast<const unsigned*>(base + 2 * D_);
        *reinterpret_cast<unsigned*>(&ldsQ[p * 32 + d2]) = qv;
        *reinterpret_cast<unsigned*>(&ldsK[p * 32 + d2]) = kv;
        bf16 v0 = reinterpret_cast<const bf16*>(&vv)[0];
        bf16 v1 = reinterpret_cast<const bf16*>(&vv)[1];
        ldsVt[(d2)     * 64 + p] = v0;
        ldsVt[(d2 + 1) * 64 + p] = v1;
    }
    __syncthreads();

    // S strip: wave w owns rows 16w..16w+15
    v16bf aq = load_a_frag(ldsQ, 16 * w, 32, 0, lane);
    v8f s[4];
#pragma unroll
    for (int j = 0; j < 4; ++j) {
        v16bf bk = load_bT_frag(ldsK, 16 * j, 32, 0, lane);
        v8f z = {};
        s[j] = __builtin_amdgcn_wmma_f32_16x16x32_bf16(false, aq, false, bk,
                                                       (short)0, z, false, false);
    }

    const float scale = 0.17677669529663687f;   // 1/sqrt(32)
    const float* relb = relh + (((size_t)b * H_ + h) << 12);
    const int half = lane >> 4, cl = lane & 15;

#pragma unroll
    for (int r = 0; r < 8; ++r) {
        int row  = 16 * w + r + 8 * half;
        float mx = -1e30f;
#pragma unroll
        for (int j = 0; j < 4; ++j) {
            float v = s[j][r] * scale + relb[row * 64 + 16 * j + cl];
            s[j][r] = v;
            mx = fmaxf(mx, v);
        }
        for (int m = 1; m < 16; m <<= 1) mx = fmaxf(mx, __shfl_xor(mx, m, 32));
        float sum = 0.f;
#pragma unroll
        for (int j = 0; j < 4; ++j) {
            float e = __expf(s[j][r] - mx);
            s[j][r] = e;
            sum += e;
        }
        for (int m = 1; m < 16; m <<= 1) sum += __shfl_xor(sum, m, 32);
        float inv = 1.0f / sum;
#pragma unroll
        for (int j = 0; j < 4; ++j)
            ldsP[row * 64 + 16 * j + cl] = (bf16)(s[j][r] * inv);
    }
    __syncthreads();

    // O strip = P(16x64) * V(64x32)
    v8f oacc[2] = {};
#pragma unroll
    for (int kk = 0; kk < 2; ++kk) {
        v16bf ap = load_a_frag(ldsP, 16 * w, 64, kk * 32, lane);
#pragma unroll
        for (int nt = 0; nt < 2; ++nt) {
            v16bf bv = load_bT_frag(ldsVt, 16 * nt, 64, kk * 32, lane);
            oacc[nt] = __builtin_amdgcn_wmma_f32_16x16x32_bf16(
                false, ap, false, bv, (short)0, oacc[nt], false, false);
        }
    }
#pragma unroll
    for (int nt = 0; nt < 2; ++nt)
#pragma unroll
        for (int r = 0; r < 8; ++r) {
            int p  = 16 * w + r + 8 * half;
            int dk = 16 * nt + cl;
            o[(rowBase + p) * D_ + off + dk] = (bf16)oacc[nt][r];
        }
}

// ---------------------------------------------------------------------------
// Host-side launch
// ---------------------------------------------------------------------------
extern "C" void kernel_launch(void* const* d_in, const int* in_sizes, int n_in,
                              void* d_out, int out_size, void* d_ws, size_t ws_size,
                              hipStream_t stream) {
    (void)in_sizes; (void)n_in; (void)out_size; (void)ws_size;
    const float* x     = (const float*)d_in[0];
    const float* pos   = (const float*)d_in[1];
    const float* w_qkv = (const float*)d_in[2];
    const float* b_qkv = (const float*)d_in[3];
    const float* w_out = (const float*)d_in[4];
    const float* b_out = (const float*)d_in[5];
    const float* w_ff1 = (const float*)d_in[6];
    const float* b_ff1 = (const float*)d_in[7];
    const float* w_ff2 = (const float*)d_in[8];
    const float* b_ff2 = (const float*)d_in[9];
    const float* w_rel = (const float*)d_in[10];
    const float* g1    = (const float*)d_in[11];
    const float* be1   = (const float*)d_in[12];
    const float* g2    = (const float*)d_in[13];
    const float* be2   = (const float*)d_in[14];

    char* ws = (char*)d_ws;
    const size_t OFF1 = (size_t)NROWS * 768 * 2;           // qkv bf16 (402.7 MB)
    const size_t OFF2 = OFF1 + (size_t)NROWS * D_ * 2;     // xn bf16  (134.2 MB)
    const size_t OFF3 = OFF2 + (size_t)NROWS * FF_ * 2;    // o/h bf16 (268.4 MB)
    const size_t OFF4 = OFF3 + (size_t)B_ * H_ * P_ * P_ * 4;

    bf16*  qkv_bf = (bf16*)ws;
    float* xt2    = (float*)ws;             // reuse qkv region after attention
    bf16*  xn_bf  = (bf16*)(ws + OFF1);
    bf16*  y_bf   = xn_bf;                  // reuse after QKV GEMM
    bf16*  o_bf   = (bf16*)(ws + OFF2);
    bf16*  h_bf   = o_bf;                   // reuse after out-proj
    float* relh   = (float*)(ws + OFF3);
    bf16*  wq     = (bf16*)(ws + OFF4);     // all weights stored TRANSPOSED (N x K)
    bf16*  wo     = wq  + D_ * 3 * D_;
    bf16*  wf1    = wo  + D_ * D_;
    bf16*  wf2    = wf1 + D_ * FF_;

    cvt_transpose_kernel<<<(D_ * 3 * D_ + 255) / 256, 256, 0, stream>>>(w_qkv, wq, D_, 3 * D_);
    cvt_transpose_kernel<<<(D_ * D_ + 255) / 256,     256, 0, stream>>>(w_out, wo, D_, D_);
    cvt_transpose_kernel<<<(D_ * FF_ + 255) / 256,    256, 0, stream>>>(w_ff1, wf1, D_, FF_);
    cvt_transpose_kernel<<<(FF_ * D_ + 255) / 256,    256, 0, stream>>>(w_ff2, wf2, FF_, D_);

    relpos_kernel<<<(B_ * P_ * P_) / 256, 256, 0, stream>>>(pos, w_rel, relh);

    ln1_kernel<<<NROWS / 8, 256, 0, stream>>>(x, g1, be1, xn_bf);

    gemm_bf16_kernel<0><<<dim3(768 / 128, NROWS / 128), 256, 0, stream>>>(
        xn_bf, wq, b_qkv, NROWS, 768, 256, nullptr, nullptr, qkv_bf);

    attn_kernel<<<B_ * T_ * H_, 128, 0, stream>>>(qkv_bf, relh, o_bf);

    gemm_bf16_kernel<2><<<dim3(256 / 128, NROWS / 128), 256, 0, stream>>>(
        o_bf, wo, b_out, NROWS, 256, 256, x, xt2, nullptr);

    ln2_kernel<<<NROWS / 8, 256, 0, stream>>>(xt2, g2, be2, y_bf);

    gemm_bf16_kernel<1><<<dim3(512 / 128, NROWS / 128), 256, 0, stream>>>(
        y_bf, wf1, b_ff1, NROWS, 512, 256, nullptr, nullptr, h_bf);

    gemm_bf16_kernel<3><<<dim3(256 / 128, NROWS / 128), 256, 0, stream>>>(
        h_bf, wf2, b_ff2, NROWS, 256, 512, xt2, (float*)d_out, nullptr);
}
